// TOGL_8340826489356
// MI455X (gfx1250) — matrix operand order repaired and checked by reading, not compiled, not run on hardware
//
#include <hip/hip_runtime.h>

#define N_NODES 4096
#define N_EDGES 65536
#define N_FEAT  128
#define HIDDEN  64
#define N_FILT  8

typedef __attribute__((ext_vector_type(16))) _Float16 v16h;
typedef __attribute__((ext_vector_type(8)))  _Float16 v8h;
typedef __attribute__((ext_vector_type(8)))  float    v8f;
typedef __attribute__((ext_vector_type(4)))  unsigned int u32x4;
typedef __attribute__((ext_vector_type(8)))  int      i32x8;
typedef __attribute__((ext_vector_type(4)))  int      i32x4;

#define W1P_ELEMS (4 * 4 * 32 * 16)   /* 8192 f16: [kk=4][nt=4][lane=32][16] */
#define W2P_ELEMS (2 * 32 * 16)       /* 1024 f16: [kk=2][lane=32][16]       */

// ---------------------------------------------------------------------------
// TDM: 1-D f16 block copy global -> LDS (Tensor DMA descriptor per ISA ch.8).
// Group0: count=1 | lds_addr | global_addr[56:0] | type=2 ("image").
// Group1: data_size=1 (2 bytes), tensor_dim0=n, tensor_dim1=1, tile_dim0=n,
//         tensor_dim0_stride=n; groups 2/3 zero (dims 2+ unused).
// 6-arg builtin variant (clang-23/therock headers): (g0, g1, g2, g3, g4, cpol).
// ---------------------------------------------------------------------------
__device__ inline void tdm_load_1d_f16(const _Float16* gsrc, _Float16* lds_dst,
                                       unsigned int nelem) {
  unsigned long long ga = (unsigned long long)(uintptr_t)gsrc;
  unsigned int lds = (unsigned int)(uintptr_t)lds_dst;
  u32x4 g0;
  g0[0] = 1u;                                              // count=1 (user D#)
  g0[1] = lds;                                             // lds_addr [63:32]
  g0[2] = (unsigned int)(ga & 0xFFFFFFFFu);                // global_addr lo
  g0[3] = (unsigned int)((ga >> 32) & 0x1FFFFFFu) | 0x80000000u; // hi | type=2
  i32x8 g1;
  g1[0] = 0x00010000;                                      // data_size=1 (2B)
  g1[1] = (int)((nelem & 0xFFFFu) << 16);                  // tensor_dim0[15:0]
  g1[2] = (int)(((nelem >> 16) & 0xFFFFu) | (1u << 16));   // dim0 hi | dim1=1
  g1[3] = (int)((nelem & 0xFFFFu) << 16);                  // tile_dim0
  g1[4] = 0;                                               // tile_dim1/2 = 0
  g1[5] = (int)nelem;                                      // dim0_stride lo
  g1[6] = 0;
  g1[7] = 0;
  i32x4 z4 = {0, 0, 0, 0};
  i32x8 z8 = {0, 0, 0, 0, 0, 0, 0, 0};
  __builtin_amdgcn_tensor_load_to_lds(g0, g1, z4, z4, z8, 0);
}

// ---------------------------------------------------------------------------
// Stage 0a: X (f32) -> Xh (f16), row-major.
// ---------------------------------------------------------------------------
__global__ __launch_bounds__(256)
void convert_x_kernel(const float* __restrict__ X, _Float16* __restrict__ Xh) {
  int i = blockIdx.x * blockDim.x + threadIdx.x;
  if (i < N_NODES * N_FEAT) Xh[i] = (_Float16)X[i];
}

// ---------------------------------------------------------------------------
// Stage 0b: pack W1/W2 into WMMA B-fragment order so each lane's 16 f16 are
// contiguous (32B). B-frag (16-bit, 32x16): lane holds col N=lane&15,
// K = kk*32 + (lane>>4)*8 + (i&7) + (i>>3)*16.
// ---------------------------------------------------------------------------
__global__ __launch_bounds__(256)
void pack_w_kernel(const float* __restrict__ W1, const float* __restrict__ W2,
                   _Float16* __restrict__ W1p, _Float16* __restrict__ W2p) {
  int idx = blockIdx.x * blockDim.x + threadIdx.x;
  if (idx < W1P_ELEMS) {
    int i = idx & 15, lane = (idx >> 4) & 31, t = idx >> 9; // t = kk*4+nt
    int kk = t >> 2, nt = t & 3;
    int k = kk * 32 + (lane >> 4) * 8 + (i & 7) + (i >> 3) * 16;
    int col = nt * 16 + (lane & 15);
    W1p[idx] = (_Float16)W1[k * HIDDEN + col];
  } else if (idx < W1P_ELEMS + W2P_ELEMS) {
    int idx2 = idx - W1P_ELEMS;
    int i = idx2 & 15, lane = (idx2 >> 4) & 31, kk = idx2 >> 9;
    int k = kk * 32 + (lane >> 4) * 8 + (i & 7) + (i >> 3) * 16;
    int col = lane & 15;
    W2p[idx2] = (col < N_FILT) ? (_Float16)W2[k * N_FILT + col] : (_Float16)0.0f;
  }
}

// ---------------------------------------------------------------------------
// Stage 1: filt = ReLU(Xh @ W1 + b1) @ W2 + b2 via v_wmma_f32_16x16x32_f16.
// One wave per 16-row tile; packed weights staged into LDS by the TDM and
// consumed as ds_load_b128 fragments. A-frags are two global b128 loads.
// ---------------------------------------------------------------------------
__global__ __launch_bounds__(32)
void mlp_filt_kernel(const _Float16* __restrict__ Xh,
                     const _Float16* __restrict__ W1p,
                     const _Float16* __restrict__ W2p,
                     const float* __restrict__ b1, const float* __restrict__ b2,
                     float* __restrict__ filt) {
  __shared__ _Float16 W1s[W1P_ELEMS];
  __shared__ _Float16 W2s[W2P_ELEMS];
  __shared__ _Float16 Hs[16 * HIDDEN];

  const int lane = threadIdx.x;
  const int rowBase = blockIdx.x * 16;
  const int half = lane >> 4;
  const int l15  = lane & 15;

  // Kick off async weight staging (TENSORcnt) ...
  tdm_load_1d_f16(W1p, W1s, W1P_ELEMS);
  tdm_load_1d_f16(W2p, W2s, W2P_ELEMS);

  // ... overlap with A-fragment loads from global.
  v16h afr[N_FEAT / 32];
#pragma unroll
  for (int kk = 0; kk < N_FEAT / 32; ++kk) {
    const _Float16* p = Xh + (size_t)(rowBase + l15) * N_FEAT + kk * 32 + half * 8;
    v8h lo = *(const v8h*)p;
    v8h hi = *(const v8h*)(p + 16);
#pragma unroll
    for (int i = 0; i < 8; ++i) { afr[kk][i] = lo[i]; afr[kk][8 + i] = hi[i]; }
  }

  __builtin_amdgcn_s_wait_tensorcnt(0);
  __syncthreads();

  // GEMM1: [16 x 128] x [128 x 64]
  v8f acc[4] = {};
#pragma unroll
  for (int kk = 0; kk < N_FEAT / 32; ++kk) {
#pragma unroll
    for (int nt = 0; nt < 4; ++nt) {
      v16h b = *(const v16h*)&W1s[((kk * 4 + nt) * 32 + lane) * 16];
      acc[nt] = __builtin_amdgcn_wmma_f32_16x16x32_f16(
          false, afr[kk], false, b, (short)0, acc[nt], false, false);
    }
  }

  // bias + ReLU -> LDS (row-major 16 x 64, f16)
#pragma unroll
  for (int nt = 0; nt < 4; ++nt) {
#pragma unroll
    for (int r = 0; r < 8; ++r) {
      const int m = r + 8 * half;
      const int c = nt * 16 + l15;
      float hv = acc[nt][r] + b1[c];
      hv = hv > 0.0f ? hv : 0.0f;
      Hs[m * HIDDEN + c] = (_Float16)hv;
    }
  }
  __syncthreads();

  // GEMM2: [16 x 64] x [64 x 16(pad of 8)]
  v8f acc2 = {};
#pragma unroll
  for (int kk = 0; kk < HIDDEN / 32; ++kk) {
    v16h a;
    {
      const _Float16* p = Hs + l15 * HIDDEN + kk * 32 + half * 8;
      v8h lo = *(const v8h*)p;
      v8h hi = *(const v8h*)(p + 16);
#pragma unroll
      for (int i = 0; i < 8; ++i) { a[i] = lo[i]; a[8 + i] = hi[i]; }
    }
    v16h b = *(const v16h*)&W2s[(kk * 32 + lane) * 16];
    acc2 = __builtin_amdgcn_wmma_f32_16x16x32_f16(
        false, a, false, b, (short)0, acc2, false, false);
  }

#pragma unroll
  for (int r = 0; r < 8; ++r) {
    const int m = r + 8 * half;
    const int c = l15;
    if (c < N_FILT)
      filt[(size_t)(rowBase + m) * N_FILT + c] = acc2[r] + b2[c];
  }
}

// ---------------------------------------------------------------------------
// Stage 2: CSR of in-edges (tgt -> list of src)
// ---------------------------------------------------------------------------
__global__ __launch_bounds__(256)
void zero_kernel(int* __restrict__ p, int n) {
  int i = blockIdx.x * blockDim.x + threadIdx.x;
  if (i < n) p[i] = 0;
}

__global__ __launch_bounds__(256)
void hist_kernel(const int* __restrict__ edges, int* __restrict__ counts) {
  int e = blockIdx.x * blockDim.x + threadIdx.x;
  if (e < N_EDGES) atomicAdd(&counts[edges[N_EDGES + e]], 1);
}

__global__ __launch_bounds__(256)
void scan_kernel(const int* __restrict__ counts, int* __restrict__ rowptr,
                 int* __restrict__ cursor) {
  __shared__ int partial[256];
  __shared__ int partial_scan[257];
  const int tid = threadIdx.x;
  const int base = tid * 16;
  int local[16];
  int s = 0;
#pragma unroll
  for (int i = 0; i < 16; ++i) { local[i] = s; s += counts[base + i]; }
  partial[tid] = s;
  __syncthreads();
  if (tid == 0) {
    int acc = 0;
    for (int t = 0; t < 256; ++t) { partial_scan[t] = acc; acc += partial[t]; }
    partial_scan[256] = acc;
  }
  __syncthreads();
  const int off = partial_scan[tid];
#pragma unroll
  for (int i = 0; i < 16; ++i) {
    int v = off + local[i];
    rowptr[base + i] = v;
    cursor[base + i] = v;
  }
  if (tid == 0) rowptr[N_NODES] = partial_scan[256];
}

__global__ __launch_bounds__(256)
void fill_kernel(const int* __restrict__ edges, int* __restrict__ cursor,
                 int* __restrict__ csr_src) {
  int e = blockIdx.x * blockDim.x + threadIdx.x;
  if (e < N_EDGES) {
    int src = edges[e];
    int tgt = edges[N_EDGES + e];
    int pos = atomicAdd(&cursor[tgt], 1);
    csr_src[pos] = src;
  }
}

// ---------------------------------------------------------------------------
// Stage 3: stable ascending sort per filtration (bitonic, tie-break on index)
// ---------------------------------------------------------------------------
__global__ __launch_bounds__(256)
void sort_kernel(const float* __restrict__ filt, int* __restrict__ order_g,
                 int* __restrict__ rank_g) {
  __shared__ float keyS[N_NODES];
  __shared__ int   idxS[N_NODES];
  const int f = blockIdx.x;
  for (int i = threadIdx.x; i < N_NODES; i += 256) {
    keyS[i] = filt[(size_t)i * N_FILT + f];
    idxS[i] = i;
  }
  __syncthreads();
  for (int k = 2; k <= N_NODES; k <<= 1) {
    for (int j = k >> 1; j > 0; j >>= 1) {
      for (int i = threadIdx.x; i < N_NODES; i += 256) {
        int ixj = i ^ j;
        if (ixj > i) {
          bool up = ((i & k) == 0);
          float ka = keyS[i], kb = keyS[ixj];
          int   ia = idxS[i], ib = idxS[ixj];
          bool agt = (ka > kb) || (ka == kb && ia > ib);
          if (agt == up) {
            keyS[i] = kb; keyS[ixj] = ka;
            idxS[i] = ib; idxS[ixj] = ia;
          }
        }
      }
      __syncthreads();
    }
  }
  for (int i = threadIdx.x; i < N_NODES; i += 256) {
    int node = idxS[i];
    order_g[f * N_NODES + i] = node;
    rank_g[f * N_NODES + node] = i;
  }
}

// ---------------------------------------------------------------------------
// Stage 4: incremental union-find persistence, one wave32 block per
// filtration; parent/death/flags LDS-resident (48 KB << 320 KB / WGP).
// ---------------------------------------------------------------------------
__global__ __launch_bounds__(32)
void persist_kernel(const int* __restrict__ order_g, const int* __restrict__ rank_g,
                    const int* __restrict__ rowptr, const int* __restrict__ csr_src,
                    float* __restrict__ out) {
  __shared__ int   parent[N_NODES];
  __shared__ float death[N_NODES];
  __shared__ int   flags[N_NODES];  // bit0 = alive, bit1 = born
  const int f = blockIdx.x;
  const int lane = threadIdx.x;
  const int* order = order_g + f * N_NODES;
  const int* rank  = rank_g  + f * N_NODES;

  for (int i = lane; i < N_NODES; i += 32) {
    parent[i] = i;
    death[i]  = __builtin_inff();
    flags[i]  = 0;
  }
  __syncthreads();
  volatile int* par = parent;

  for (int j = 0; j < N_NODES; ++j) {
    const int v = order[j];
    const int beg = rowptr[v], end = rowptr[v + 1];

    int m = j;
    for (int e = beg + lane; e < end; e += 32) {
      int ru = rank[csr_src[e]];
      if (ru < j) {
        int x = ru;
        while (true) {
          int p = par[x];
          if (p == x) break;
          int gp = par[p];
          par[x] = gp;
          x = gp;
        }
        if (x < m) m = x;
      }
    }
#pragma unroll
    for (int off = 16; off > 0; off >>= 1) {
      int o = __shfl_xor(m, off, 32);
      if (o < m) m = o;
    }
    __syncthreads();

    for (int e = beg + lane; e < end; e += 32) {
      int ru = rank[csr_src[e]];
      if (ru < j) {
        int x = ru;
        while (true) {
          int p = par[x];
          if (p == x) break;
          int gp = par[p];
          par[x] = gp;
          x = gp;
        }
        if (x != m) {
          if (flags[x] & 1) {
            death[x] = (float)(j - 1);
            flags[x] &= ~1;
          }
          par[x] = m;
        }
      }
    }
    if (lane == 0) {
      if (m == j) flags[j] = 3;
      else        par[j] = m;
    }
    __syncthreads();
  }

  float* pairs = out;
  float* mask  = out + (size_t)N_FILT * N_NODES * 2;
  for (int r = lane; r < N_NODES; r += 32) {
    pairs[((size_t)f * N_NODES + r) * 2 + 0] = (float)r;
    pairs[((size_t)f * N_NODES + r) * 2 + 1] = death[r];
    mask[(size_t)f * N_NODES + r] = (flags[r] & 2) ? 1.0f : 0.0f;
  }
}

// ---------------------------------------------------------------------------
extern "C" void kernel_launch(void* const* d_in, const int* in_sizes, int n_in,
                              void* d_out, int out_size, void* d_ws, size_t ws_size,
                              hipStream_t stream) {
  (void)in_sizes; (void)n_in; (void)out_size; (void)ws_size;
  const float* X     = (const float*)d_in[0];
  const int*   edges = (const int*)d_in[1];
  const float* W1    = (const float*)d_in[2];
  const float* b1    = (const float*)d_in[3];
  const float* W2    = (const float*)d_in[4];
  const float* b2    = (const float*)d_in[5];
  float* out = (float*)d_out;

  char* ws = (char*)d_ws;
  size_t off = 0;
  float*    filt    = (float*)(ws + off);    off += (size_t)N_NODES * N_FILT * sizeof(float);
  int*      counts  = (int*)(ws + off);      off += (size_t)(N_NODES + 1) * sizeof(int);
  int*      rowptr  = (int*)(ws + off);      off += (size_t)(N_NODES + 1) * sizeof(int);
  int*      cursor  = (int*)(ws + off);      off += (size_t)(N_NODES + 1) * sizeof(int);
  int*      csr_src = (int*)(ws + off);      off += (size_t)N_EDGES * sizeof(int);
  int*      order_g = (int*)(ws + off);      off += (size_t)N_FILT * N_NODES * sizeof(int);
  int*      rank_g  = (int*)(ws + off);      off += (size_t)N_FILT * N_NODES * sizeof(int);
  off = (off + 255) & ~(size_t)255;
  _Float16* Xh      = (_Float16*)(ws + off); off += (size_t)N_NODES * N_FEAT * sizeof(_Float16);
  _Float16* W1p     = (_Float16*)(ws + off); off += (size_t)W1P_ELEMS * sizeof(_Float16);
  _Float16* W2p     = (_Float16*)(ws + off); off += (size_t)W2P_ELEMS * sizeof(_Float16);

  convert_x_kernel<<<(N_NODES * N_FEAT) / 256, 256, 0, stream>>>(X, Xh);
  pack_w_kernel<<<(W1P_ELEMS + W2P_ELEMS + 255) / 256, 256, 0, stream>>>(W1, W2, W1p, W2p);
  mlp_filt_kernel<<<N_NODES / 16, 32, 0, stream>>>(Xh, W1p, W2p, b1, b2, filt);
  zero_kernel<<<(N_NODES + 1 + 255) / 256, 256, 0, stream>>>(counts, N_NODES + 1);
  hist_kernel<<<N_EDGES / 256, 256, 0, stream>>>(edges, counts);
  scan_kernel<<<1, 256, 0, stream>>>(counts, rowptr, cursor);
  fill_kernel<<<N_EDGES / 256, 256, 0, stream>>>(edges, cursor, csr_src);
  sort_kernel<<<N_FILT, 256, 0, stream>>>(filt, order_g, rank_g);
  persist_kernel<<<N_FILT, 32, 0, stream>>>(order_g, rank_g, rowptr, csr_src, out);
}
